// MSGACD_Unmixer_15556371546821
// MI455X (gfx1250) — compile-verified, matrix operand order checked by reading, not run on hardware
//
// MI455X (gfx1250) implementation of the MSGACD unmixer reference.
// Heavy GEMM phases (DiffSLIC distances, Q^T x center update, 3x3 convs as
// implicit GEMM) use v_wmma_f32_16x16x32_bf16 (wave32). Transposed-fragment
// GEMMs stage tiles into LDS via the Tensor Data Mover (tensor_load_to_lds +
// s_wait_tensorcnt) when the builtin exists, else cooperative b128 copies.
// Small phases (GAT on 120 nodes, fusion, PCR, decode) are VALU kernels.
#include <hip/hip_runtime.h>

#define BB 2
#define CC 200
#define CP 224      // CC padded to multiple of 32
#define HH 256
#define WW 256
#define NN 65536    // HH*WW
#define PP 12
#define KSP 120
#define KP 128      // KSP padded to multiple of 16
#define GH 64
#define NSEG 1000

typedef float  v8f   __attribute__((ext_vector_type(8)));
typedef __bf16 v16bf __attribute__((ext_vector_type(16)));
typedef unsigned int u32x4 __attribute__((ext_vector_type(4)));
typedef int i32x4 __attribute__((ext_vector_type(4)));
typedef int i32x8 __attribute__((ext_vector_type(8)));

#if __has_builtin(__builtin_amdgcn_tensor_load_to_lds)
#define HAVE_TDM 1
#endif

// ---------------------------------------------------------------- TDM
// 2D tile load into LDS: rows x rowElems of 2-byte elements, contiguous rows
// (tensor_dim0_stride == rowElems). D# per ISA 08_async_tensor ยง8.3/8.4.
static __device__ inline void tdm_load_tile_2d(unsigned lds_off, const void* g,
                                               unsigned rowElems, unsigned rows) {
#ifdef HAVE_TDM
  unsigned long long ga = (unsigned long long)g;
  u32x4 g0;
  g0[0] = 1u;                                   // count=1, user flags 0
  g0[1] = lds_off;                              // lds_addr (bytes)
  g0[2] = (unsigned)ga;                         // global_addr[31:0]
  g0[3] = (unsigned)((ga >> 32) & 0x1FFFFFFu) | 0x80000000u;  // [56:32] | type=2
  i32x8 g1;
  g1[0] = (int)(1u << 16);                      // workgroup_mask=0, data_size=1 (2B)
  g1[1] = (int)((rowElems & 0xFFFFu) << 16);    // tensor_dim0[15:0]
  g1[2] = (int)(((rowElems >> 16) & 0xFFFFu) |  // tensor_dim0[31:16]
                (((unsigned)NN & 0xFFFFu) << 16));  // tensor_dim1[15:0]
  g1[3] = (int)((((unsigned)NN >> 16) & 0xFFFFu) |  // tensor_dim1[31:16]
                ((rowElems & 0xFFFFu) << 16));  // tile_dim0
  g1[4] = (int)(rows & 0xFFFFu);                // tile_dim1 (tile_dim2 = 0)
  g1[5] = (int)rowElems;                        // tensor_dim0_stride[31:0]
  g1[6] = 0;                                    // stride0 hi, stride1 lo
  g1[7] = 0;
  i32x4 z4 = {0, 0, 0, 0};
#if __clang_major__ >= 23
  i32x8 z8 = {0, 0, 0, 0, 0, 0, 0, 0};
  __builtin_amdgcn_tensor_load_to_lds(g0, g1, z4, z4, z8, 0);
#else
  __builtin_amdgcn_tensor_load_to_lds(g0, g1, z4, z4, 0);
#endif
#else
  (void)lds_off; (void)g; (void)rowElems; (void)rows;
#endif
}

// ---------------------------------------------------------------- fragments
// 16-bit A-matrix 16x32 layout (ISA 7.12.2): lane holds row M = lane&15;
// lanes 0-15: K = {0..7, 16..23}, lanes 16-31: K = {8..15, 24..31}.
// B (32x16) mirrors this with col = lane&15.
static __device__ inline v16bf frag_zero() {
  v16bf f;
#pragma unroll
  for (int i = 0; i < 16; ++i) f[i] = (__bf16)0.0f;
  return f;
}

// p points at (this lane's row/col, k=0); ks = element stride along K.
static __device__ inline v16bf frag_load(const __bf16* __restrict__ p, int ks) {
  v16bf f;
  const int hb = (threadIdx.x & 16) ? 8 : 0;
#pragma unroll
  for (int j = 0; j < 8; ++j) {
    f[j]     = p[(size_t)(hb + j) * ks];
    f[8 + j] = p[(size_t)(16 + hb + j) * ks];
  }
  return f;
}

static __device__ inline v8f wmma_bf16(v16bf a, v16bf b, v8f c) {
  return __builtin_amdgcn_wmma_f32_16x16x32_bf16(false, a, false, b, (short)0, c,
                                                 false, false);
}

// ---------------------------------------------------------------- pack x
__global__ __launch_bounds__(256) void k_pack_x(const float* __restrict__ x,
                                                __bf16* __restrict__ xT,
                                                float* __restrict__ xnorm) {
  const int n = blockIdx.x * 256 + threadIdx.x;
  const int b = blockIdx.y;
  const float* xb = x + (size_t)b * CC * NN;
  __bf16* row = xT + ((size_t)b * NN + n) * CP;
  float s = 0.f;
  for (int c = 0; c < CC; ++c) {
    float v = xb[(size_t)c * NN + n];
    s += v * v;
    row[c] = (__bf16)v;
  }
  for (int c = CC; c < CP; ++c) row[c] = (__bf16)0.0f;
  xnorm[b * NN + n] = s;
}

// ---------------------------------------------------------------- SLIC init
__global__ __launch_bounds__(256) void k_init_centers(const float* __restrict__ x,
                                                      __bf16* __restrict__ cbf,
                                                      float* __restrict__ cf32,
                                                      float* __restrict__ csp) {
  const int tid = threadIdx.x;
  for (int idx = tid; idx < KP * CP; idx += 256) {
    const int k = idx / CP, c = idx % CP;
    float v = 0.f;
    if (k < KSP && c < CC) {
      const int i = k / 12, j = k % 12;
      const int gy = 12 + 25 * i, gx = 10 + 21 * j;
      const int flat = gy * WW + gx;
      v = x[(size_t)flat * CC + c];   // reference's reshape(-1,C) quirk
    }
    for (int b = 0; b < BB; ++b) {
      cbf[((size_t)b * KP + k) * CP + c] = (__bf16)v;
      cf32[((size_t)b * KP + k) * CP + c] = v;
    }
  }
  for (int k = tid; k < KP; k += 256) {
    float gy = 0.f, gx = 0.f;
    if (k < KSP) {
      gy = (float)(12 + 25 * (k / 12));
      gx = (float)(10 + 21 * (k % 12));
    }
    for (int b = 0; b < BB; ++b) {
      csp[((size_t)b * KP + k) * 2 + 0] = gy;
      csp[((size_t)b * KP + k) * 2 + 1] = gx;
    }
  }
}

__global__ __launch_bounds__(256) void k_cnorm(const __bf16* __restrict__ cbf,
                                               float* __restrict__ cnorm) {
  const int t = threadIdx.x;
  if (t >= BB * KP) return;
  const __bf16* row = cbf + (size_t)t * CP;
  float s = 0.f;
  for (int c = 0; c < CP; ++c) { float v = (float)row[c]; s += v * v; }
  cnorm[t] = s;
}

// ---------------------------------------------------------------- SLIC assign
// dyn LDS: [0, 57344) centers bf16 128x224, then mass/y/x f32 partials.
#define ASSIGN_LDS (57344 + 3 * KP * 4)
__global__ __launch_bounds__(256) void k_assign(
    const __bf16* __restrict__ xT, const float* __restrict__ xnorm,
    const __bf16* __restrict__ cbf, const float* __restrict__ cnorm,
    const float* __restrict__ csp, __bf16* __restrict__ Q,
    float* __restrict__ mass, float* __restrict__ csp_acc) {
  extern __shared__ char smem[];
  __bf16* cL = (__bf16*)smem;
  float* mL = (float*)(smem + 57344);
  float* yL = mL + KP;
  float* xL = mL + 2 * KP;

  const int b = blockIdx.y;
  const int wave = threadIdx.x >> 5;
  const int lane = threadIdx.x & 31;
  const int col = lane & 15;
  const int mofs = (lane & 16) ? 8 : 0;
  const int pixT = blockIdx.x * 128 + wave * 16;

  // stage centers tile (128x224 bf16) into LDS
#ifdef HAVE_TDM
  if (threadIdx.x < 32) {
    tdm_load_tile_2d(0, cbf + (size_t)b * KP * CP, CP, KP);
#if __has_builtin(__builtin_amdgcn_s_wait_tensorcnt)
    __builtin_amdgcn_s_wait_tensorcnt(0);
#endif
  }
#else
  {
    const uint4* src = (const uint4*)(cbf + (size_t)b * KP * CP);
    uint4* dst = (uint4*)cL;
    for (int i = threadIdx.x; i < (KP * CP * 2) / 16; i += 256) dst[i] = src[i];
  }
#endif
  if (threadIdx.x < KP) {
    mL[threadIdx.x] = 0.f; yL[threadIdx.x] = 0.f; xL[threadIdx.x] = 0.f;
  }
  __syncthreads();

  v8f acc[8];
#pragma unroll
  for (int i = 0; i < 8; ++i) acc[i] = (v8f)0.f;

  const __bf16* arow = xT + ((size_t)b * NN + pixT + col) * CP;
  for (int kc = 0; kc < CP; kc += 32) {
    __builtin_prefetch(arow + kc + 32, 0, 1);
    v16bf a = frag_load(arow + kc, 1);
#pragma unroll
    for (int ct = 0; ct < 8; ++ct) {
      v16bf bf = frag_load(cL + (size_t)(ct * 16 + col) * CP + kc, 1);
      acc[ct] = wmma_bf16(a, bf, acc[ct]);
    }
  }

  int pix[8];
  float xn[8];
#pragma unroll
  for (int r = 0; r < 8; ++r) {
    pix[r] = pixT + r + mofs;
    xn[r] = xnorm[b * NN + pix[r]];
  }
  const float coef = 10.0f / sqrtf((float)NN / (float)KSP);

#pragma unroll
  for (int ct = 0; ct < 8; ++ct) {
    const int n = ct * 16 + col;
    const float cn = cnorm[b * KP + n];
    const float cy = csp[((size_t)b * KP + n) * 2 + 0];
    const float cx = csp[((size_t)b * KP + n) * 2 + 1];
    const bool ok = (n < KSP);
#pragma unroll
    for (int r = 0; r < 8; ++r) {
      float g = acc[ct][r];
      float df = sqrtf(fmaxf(xn[r] + cn - 2.f * g, 1e-12f));
      float py = (float)(pix[r] >> 8), px = (float)(pix[r] & 255);
      float ds = sqrtf(fmaxf((py - cy) * (py - cy) + (px - cx) * (px - cx), 1e-12f));
      acc[ct][r] = ok ? (df + coef * ds) : 1e30f;
    }
  }

  // softmax(-D) over 128 cols (120 valid); rows live on 16-lane halves, so
  // xor-shuffles 1,2,4,8 reduce exactly within a half.
#pragma unroll
  for (int r = 0; r < 8; ++r) {
    float m = 1e30f;
#pragma unroll
    for (int ct = 0; ct < 8; ++ct) m = fminf(m, acc[ct][r]);
    for (int o = 1; o <= 8; o <<= 1) m = fminf(m, __shfl_xor(m, o, 32));
    float s = 0.f;
#pragma unroll
    for (int ct = 0; ct < 8; ++ct) s += expf(m - acc[ct][r]);
    for (int o = 1; o <= 8; o <<= 1) s += __shfl_xor(s, o, 32);
    const float inv = 1.f / s;
#pragma unroll
    for (int ct = 0; ct < 8; ++ct) {
      float q = expf(m - acc[ct][r]) * inv;
      acc[ct][r] = q;
      Q[((size_t)b * NN + pix[r]) * KP + ct * 16 + col] = (__bf16)q;
    }
  }

#pragma unroll
  for (int ct = 0; ct < 8; ++ct) {
    const int n = ct * 16 + col;
    float pm = 0.f, py = 0.f, px = 0.f;
#pragma unroll
    for (int r = 0; r < 8; ++r) {
      float q = acc[ct][r];
      pm += q;
      py += q * (float)(pix[r] >> 8);
      px += q * (float)(pix[r] & 255);
    }
    atomicAdd(&mL[n], pm);
    atomicAdd(&yL[n], py);
    atomicAdd(&xL[n], px);
  }
  __syncthreads();
  if (threadIdx.x < KP) {
    atomicAdd(&mass[b * KP + threadIdx.x], mL[threadIdx.x]);
    atomicAdd(&csp_acc[((size_t)b * KP + threadIdx.x) * 2 + 0], yL[threadIdx.x]);
    atomicAdd(&csp_acc[((size_t)b * KP + threadIdx.x) * 2 + 1], xL[threadIdx.x]);
  }
}

// ---------------------------------------------------------------- SLIC update
// cs_acc[k][c] += sum_pix Q[pix][k] * x[pix][c]. Transposed fragments, so each
// 32-pixel step of Q (8KB) and xT (14KB) is staged to LDS (TDM if available)
// and the strided 2-byte gathers hit LDS instead of global memory.
#define UPD_PIX 1024
#define UPD_LDS (32 * KP * 2 + 32 * CP * 2)
__global__ __launch_bounds__(256) void k_update(const __bf16* __restrict__ Q,
                                                const __bf16* __restrict__ xT,
                                                float* __restrict__ cs_acc) {
  extern __shared__ char smem[];
  __bf16* qL = (__bf16*)smem;                 // 32 x 128
  __bf16* xL = (__bf16*)(smem + 32 * KP * 2); // 32 x 224

  const int b = blockIdx.y;
  const int wave = threadIdx.x >> 5;  // wave == k-tile (8 tiles = 128 centers)
  const int lane = threadIdx.x & 31;
  const int col = lane & 15;
  const int mofs = (lane & 16) ? 8 : 0;
  const int p0base = blockIdx.x * UPD_PIX;

  v8f acc[14];
#pragma unroll
  for (int i = 0; i < 14; ++i) acc[i] = (v8f)0.f;

  for (int p0 = p0base; p0 < p0base + UPD_PIX; p0 += 32) {
#ifdef HAVE_TDM
    if (threadIdx.x < 32) {
      tdm_load_tile_2d(0, Q + ((size_t)b * NN + p0) * KP, KP, 32);
      tdm_load_tile_2d(32 * KP * 2, xT + ((size_t)b * NN + p0) * CP, CP, 32);
#if __has_builtin(__builtin_amdgcn_s_wait_tensorcnt)
      __builtin_amdgcn_s_wait_tensorcnt(0);
#endif
    }
#else
    {
      const uint4* sq = (const uint4*)(Q + ((size_t)b * NN + p0) * KP);
      uint4* dq = (uint4*)qL;
      for (int i = threadIdx.x; i < (32 * KP * 2) / 16; i += 256) dq[i] = sq[i];
      const uint4* sx = (const uint4*)(xT + ((size_t)b * NN + p0) * CP);
      uint4* dx = (uint4*)xL;
      for (int i = threadIdx.x; i < (32 * CP * 2) / 16; i += 256) dx[i] = sx[i];
    }
#endif
    __syncthreads();
    // A = Q^T : element (row=k, kk=pixel) at qL[kk*KP + k]
    v16bf a = frag_load(qL + wave * 16 + col, KP);
#pragma unroll
    for (int ct = 0; ct < 14; ++ct) {
      // B : element (kk=pixel, col=c) at xL[kk*CP + c]
      v16bf bf = frag_load(xL + ct * 16 + col, CP);
      acc[ct] = wmma_bf16(a, bf, acc[ct]);
    }
    __syncthreads();
  }
#pragma unroll
  for (int ct = 0; ct < 14; ++ct)
#pragma unroll
    for (int r = 0; r < 8; ++r)
      atomicAdd(&cs_acc[((size_t)b * KP + wave * 16 + r + mofs) * CP + ct * 16 + col],
                acc[ct][r]);
}

__global__ __launch_bounds__(256) void k_finalize(
    const float* __restrict__ cs_acc, const float* __restrict__ csp_acc,
    const float* __restrict__ mass, __bf16* __restrict__ cbf,
    float* __restrict__ cf32, float* __restrict__ cnorm, float* __restrict__ csp) {
  const int bk = blockIdx.x;               // b*KP + k
  const int tid = threadIdx.x;
  const float mm = mass[bk] + 1e-6f;
  __shared__ float red[256];
  float v = 0.f;
  if (tid < CP) {
    float c = cs_acc[(size_t)bk * CP + tid] / mm;
    __bf16 cb = (__bf16)c;
    cbf[(size_t)bk * CP + tid] = cb;
    cf32[(size_t)bk * CP + tid] = c;
    float cbv = (float)cb;
    v = cbv * cbv;
  }
  red[tid] = v;
  __syncthreads();
  for (int s = 128; s > 0; s >>= 1) {
    if (tid < s) red[tid] += red[tid + s];
    __syncthreads();
  }
  if (tid == 0) cnorm[bk] = red[0];
  if (tid < 2) csp[(size_t)bk * 2 + tid] = csp_acc[(size_t)bk * 2 + tid] / mm;
}

// ---------------------------------------------------------------- GAT (tiny)
__global__ __launch_bounds__(256) void k_gat(
    const float* __restrict__ cf32, const float* __restrict__ g1W,
    const float* __restrict__ g1a, const float* __restrict__ g2W,
    const float* __restrict__ g2a, float* __restrict__ gout) {
  const int b = blockIdx.x;
  const int tid = threadIdx.x;
  __shared__ float Wh[KSP * GH];
  __shared__ float Hh[KSP * GH];
  __shared__ float es[KSP], ed[KSP];

  for (int idx = tid; idx < KSP * GH; idx += 256) {
    const int k = idx / GH, f = idx % GH;
    const float* feat = cf32 + ((size_t)b * KP + k) * CP;
    const float* wr = g1W + (size_t)f * CC;
    float a = 0.f;
    for (int c = 0; c < CC; ++c) a += feat[c] * wr[c];
    Wh[idx] = a;
  }
  __syncthreads();
  if (tid < KSP) {
    float a = 0.f, d = 0.f;
    for (int f = 0; f < GH; ++f) {
      a += Wh[tid * GH + f] * g1a[f];
      d += Wh[tid * GH + f] * g1a[GH + f];
    }
    es[tid] = a; ed[tid] = d;
  }
  __syncthreads();
  if (tid < KSP) {
    const float ei = es[tid];
    float m = -1e30f;
    for (int j = 0; j < KSP; ++j) {
      float e = ei + ed[j]; e = e > 0.f ? e : 0.2f * e;
      m = fmaxf(m, e);
    }
    float s = 0.f;
    for (int j = 0; j < KSP; ++j) {
      float e = ei + ed[j]; e = e > 0.f ? e : 0.2f * e;
      s += expf(e - m);
    }
    const float inv = 1.f / s;
    for (int f = 0; f < GH; ++f) {
      float a = 0.f;
      for (int j = 0; j < KSP; ++j) {
        float e = ei + ed[j]; e = e > 0.f ? e : 0.2f * e;
        a += expf(e - m) * Wh[j * GH + f];
      }
      a *= inv;
      Hh[tid * GH + f] = a > 0.f ? a : (expf(a) - 1.f);
    }
  }
  __syncthreads();
  float* Wh2 = Wh;  // reuse
  for (int idx = tid; idx < KSP * PP; idx += 256) {
    const int k = idx / PP, f = idx % PP;
    const float* wr = g2W + (size_t)f * GH;
    float a = 0.f;
    for (int c = 0; c < GH; ++c) a += Hh[k * GH + c] * wr[c];
    Wh2[idx] = a;
  }
  __syncthreads();
  if (tid < KSP) {
    float a = 0.f, d = 0.f;
    for (int f = 0; f < PP; ++f) {
      a += Wh2[tid * PP + f] * g2a[f];
      d += Wh2[tid * PP + f] * g2a[PP + f];
    }
    es[tid] = a; ed[tid] = d;
  }
  __syncthreads();
  if (tid < KSP) {
    const float ei = es[tid];
    float m = -1e30f;
    for (int j = 0; j < KSP; ++j) {
      float e = ei + ed[j]; e = e > 0.f ? e : 0.2f * e;
      m = fmaxf(m, e);
    }
    float s = 0.f;
    for (int j = 0; j < KSP; ++j) {
      float e = ei + ed[j]; e = e > 0.f ? e : 0.2f * e;
      s += expf(e - m);
    }
    const float inv = 1.f / s;
    for (int f = 0; f < PP; ++f) {
      float a = 0.f;
      for (int j = 0; j < KSP; ++j) {
        float e = ei + ed[j]; e = e > 0.f ? e : 0.2f * e;
        a += expf(e - m) * Wh2[j * PP + f];
      }
      a *= inv;
      gout[((size_t)b * KP + tid) * PP + f] = a > 0.f ? a : (expf(a) - 1.f);
    }
  } else if (tid < KP) {
    for (int f = 0; f < PP; ++f) gout[((size_t)b * KP + tid) * PP + f] = 0.f;
  }
}

// ---------------------------------------------------------------- SG = Q g
__global__ __launch_bounds__(256) void k_sg(const __bf16* __restrict__ Q,
                                            const float* __restrict__ gout,
                                            float* __restrict__ SG) {
  const int b = blockIdx.y;
  __shared__ float gL[KP * PP];
  for (int idx = threadIdx.x; idx < KP * PP; idx += 256)
    gL[idx] = gout[(size_t)b * KP * PP + idx];
  __syncthreads();
  const int n = blockIdx.x * 256 + threadIdx.x;
  const __bf16* qrow = Q + ((size_t)b * NN + n) * KP;
  float a[PP];
#pragma unroll
  for (int p = 0; p < PP; ++p) a[p] = 0.f;
  for (int k = 0; k < KP; ++k) {
    const float q = (float)qrow[k];
#pragma unroll
    for (int p = 0; p < PP; ++p) a[p] += q * gL[k * PP + p];
  }
  for (int p = 0; p < PP; ++p) SG[((size_t)b * NN + n) * PP + p] = a[p];
}

// ---------------------------------------------------------------- conv pack
__global__ __launch_bounds__(256) void k_wpack(const float* __restrict__ w,
                                               __bf16* __restrict__ wp, int O,
                                               int I, int IP) {
  const int idx = blockIdx.x * 256 + threadIdx.x;
  if (idx >= O * 9 * IP) return;
  const int o = idx / (9 * IP);
  const int rem = idx % (9 * IP);
  const int t = rem / IP, ci = rem % IP;
  float v = (ci < I) ? w[((size_t)o * I + ci) * 9 + t] : 0.f;
  wp[(size_t)o * 9 * IP + (size_t)t * IP + ci] = (__bf16)v;
}

// implicit-GEMM 3x3 conv: out[b][n][cout] = sum_tap,ci in[b][n'][ci]*w
__global__ __launch_bounds__(256) void k_conv(const __bf16* __restrict__ in,
                                              const __bf16* __restrict__ wp,
                                              float* __restrict__ out, int CinP,
                                              int Cout, int dil) {
  const int b = blockIdx.y;
  const int wave = threadIdx.x >> 5;
  const int lane = threadIdx.x & 31;
  const int col = lane & 15;
  const int mofs = (lane & 16) ? 8 : 0;
  const int pixT = blockIdx.x * 128 + wave * 16;
  const int row = lane & 15;
  const int pix = pixT + row;
  const int y = pix >> 8, xx = pix & 255;
  const int NT = Cout >> 4;
  const int ldw = 9 * CinP;

  v8f acc[8];
#pragma unroll
  for (int i = 0; i < 8; ++i) acc[i] = (v8f)0.f;

  for (int ty = 0; ty < 3; ++ty) {
    const int ys = y + (ty - 1) * dil;
    const bool vy = (unsigned)ys < (unsigned)HH;
    for (int tx = 0; tx < 3; ++tx) {
      const int xs = xx + (tx - 1) * dil;
      const bool ok = vy && ((unsigned)xs < (unsigned)WW);
      const int sp = ok ? (ys * WW + xs) : 0;
      const int t = ty * 3 + tx;
      const __bf16* ar = in + ((size_t)b * NN + sp) * CinP;
      for (int kc = 0; kc < CinP; kc += 32) {
        if (ok) __builtin_prefetch(ar + kc + 32, 0, 1);
        v16bf a = ok ? frag_load(ar + kc, 1) : frag_zero();
        for (int ct = 0; ct < NT; ++ct) {
          v16bf bf =
              frag_load(wp + (size_t)(ct * 16 + col) * ldw + (size_t)t * CinP + kc, 1);
          acc[ct] = wmma_bf16(a, bf, acc[ct]);
        }
      }
    }
  }
  for (int ct = 0; ct < NT; ++ct)
#pragma unroll
    for (int r = 0; r < 8; ++r)
      out[((size_t)b * NN + pixT + r + mofs) * Cout + ct * 16 + col] = acc[ct][r];
}

// ---------------------------------------------------------------- inst norm
__global__ __launch_bounds__(256) void k_in_stats(const float* __restrict__ v,
                                                  float* __restrict__ ms,
                                                  int Cout) {
  const int ch = blockIdx.x, b = blockIdx.y, tid = threadIdx.x;
  __shared__ float r0[256], r1[256];
  float s = 0.f, q = 0.f;
  for (int n = tid; n < NN; n += 256) {
    float x = v[((size_t)b * NN + n) * Cout + ch];
    s += x; q += x * x;
  }
  r0[tid] = s; r1[tid] = q;
  __syncthreads();
  for (int st = 128; st > 0; st >>= 1) {
    if (tid < st) { r0[tid] += r0[tid + st]; r1[tid] += r1[tid + st]; }
    __syncthreads();
  }
  if (tid == 0) {
    float mean = r0[0] / (float)NN;
    float var = r1[0] / (float)NN - mean * mean;
    ms[(b * Cout + ch) * 2 + 0] = mean;
    ms[(b * Cout + ch) * 2 + 1] = rsqrtf(var + 1e-5f);
  }
}

__global__ __launch_bounds__(256) void k_in_apply(const float* __restrict__ v,
                                                  const float* __restrict__ ms,
                                                  __bf16* __restrict__ t,
                                                  int Cout) {
  const size_t idx = (size_t)blockIdx.x * 256 + threadIdx.x;
  const int ch = (int)(idx % Cout);
  const size_t bn = idx / Cout;
  const int b = (int)(bn / NN);
  const float mean = ms[(b * Cout + ch) * 2 + 0];
  const float rstd = ms[(b * Cout + ch) * 2 + 1];
  float y = (v[idx] - mean) * rstd;
  t[idx] = (__bf16)(y > 0.f ? y : 0.f);
}

// ---------------------------------------------------------------- conv4 1x1
__global__ __launch_bounds__(256) void k_conv4(const __bf16* __restrict__ t3,
                                               const float* __restrict__ w4,
                                               const float* __restrict__ b4,
                                               float* __restrict__ SC) {
  const int b = blockIdx.y;
  __shared__ float wL[PP * 128], bL[PP];
  for (int i = threadIdx.x; i < PP * 128; i += 256) wL[i] = w4[i];
  if (threadIdx.x < PP) bL[threadIdx.x] = b4[threadIdx.x];
  __syncthreads();
  const int n = blockIdx.x * 256 + threadIdx.x;
  const __bf16* row = t3 + ((size_t)b * NN + n) * 128;
  float a[PP];
#pragma unroll
  for (int p = 0; p < PP; ++p) a[p] = bL[p];
  for (int ci = 0; ci < 128; ++ci) {
    const float v = (float)row[ci];
#pragma unroll
    for (int p = 0; p < PP; ++p) a[p] += v * wL[p * 128 + ci];
  }
  for (int p = 0; p < PP; ++p) SC[((size_t)b * NN + n) * PP + p] = a[p];
}

// ---------------------------------------------------------------- fusion
__global__ __launch_bounds__(256) void k_fusion(
    const float* __restrict__ SG, const float* __restrict__ SC,
    const float* __restrict__ fgw, const float* __restrict__ fgb,
    const float* __restrict__ flw, const float* __restrict__ flb,
    const float* __restrict__ alpha, const float* __restrict__ beta,
    float* __restrict__ S, float* __restrict__ spsum) {
  const int b = blockIdx.y;
  __shared__ float gwL[PP * 24], lwL[PP * 24], gbL[PP], lbL[PP];
  for (int i = threadIdx.x; i < PP * 24; i += 256) { gwL[i] = fgw[i]; lwL[i] = flw[i]; }
  if (threadIdx.x < PP) { gbL[threadIdx.x] = fgb[threadIdx.x]; lbL[threadIdx.x] = flb[threadIdx.x]; }
  __syncthreads();
  const float al = *alpha, be = *beta;
  const int n = blockIdx.x * 256 + threadIdx.x;
  float sg[PP], sc[PP], sv[PP];
  for (int p = 0; p < PP; ++p) {
    sg[p] = SG[((size_t)b * NN + n) * PP + p];
    sc[p] = SC[((size_t)b * NN + n) * PP + p];
  }
  for (int p = 0; p < PP; ++p) {
    float zg = gbL[p], zl = lbL[p];
    for (int q = 0; q < PP; ++q) {
      zg += gwL[p * 24 + q] * sg[q] + gwL[p * 24 + PP + q] * sc[q];
      zl += lwL[p * 24 + q] * sc[q] + lwL[p * 24 + PP + q] * sg[q];
    }
    const float FG = 1.f / (1.f + expf(-zg));
    const float FC = 1.f / (1.f + expf(-zl));
    const float sch = al * FG * sg[p] + sc[p];
    const float sgh = be * FC * sc[p] + sg[p];
    sv[p] = sgh + sch;
  }
  float m = -1e30f;
  for (int p = 0; p < PP; ++p) m = fmaxf(m, sv[p]);
  float su = 0.f;
  for (int p = 0; p < PP; ++p) su += expf(sv[p] - m);
  const float inv = 1.f / su;
  const int seg = n % NSEG;
  for (int p = 0; p < PP; ++p) {
    const float s = expf(sv[p] - m) * inv;
    S[((size_t)b * NN + n) * PP + p] = s;
    atomicAdd(&spsum[((size_t)b * NSEG + seg) * PP + p], s);
  }
}

// ---------------------------------------------------------------- PCR smooth
__global__ __launch_bounds__(256) void k_smooth(const float* __restrict__ spsum,
                                                float* __restrict__ smoothed) {
  const int b = blockIdx.x;
  __shared__ float spf[NSEG * PP];
  __shared__ float tot[PP];
  for (int idx = threadIdx.x; idx < NSEG * PP; idx += 256) {
    const int s = idx / PP;
    const float cnt = (float)(65 + (s < 536 ? 1 : 0));  // bincount(arange%1000)
    spf[idx] = spsum[(size_t)b * NSEG * PP + idx] / cnt;
  }
  __syncthreads();
  if (threadIdx.x < PP) {
    float a = 0.f;
    for (int s = 0; s < NSEG; ++s) a += spf[s * PP + threadIdx.x];
    tot[threadIdx.x] = a;
  }
  __syncthreads();
  const float e = expf(1.0f);
  const float wd = e / (e + (float)(NSEG - 1));
  const float wo = 1.f / (e + (float)(NSEG - 1));
  for (int idx = threadIdx.x; idx < NSEG * PP; idx += 256) {
    const int p = idx % PP;
    smoothed[(size_t)b * NSEG * PP + idx] = wo * tot[p] + (wd - wo) * spf[idx];
  }
}

// ---------------------------------------------------------------- endmembers
__global__ __launch_bounds__(256) void k_mc(const float* __restrict__ M,
                                            float* __restrict__ Mc) {
  __shared__ float nrm[PP];
  if (threadIdx.x < PP) {
    float a = 0.f;
    for (int c = 0; c < CC; ++c) {
      float v = M[(size_t)c * PP + threadIdx.x];
      v = v > 0.f ? v : 0.f;
      a += v * v;
    }
    nrm[threadIdx.x] = sqrtf(a) + 1e-8f;
  }
  __syncthreads();
  for (int idx = threadIdx.x; idx < CC * PP; idx += 256) {
    float v = M[idx];
    v = v > 0.f ? v : 0.f;
    Mc[idx] = v / nrm[idx % PP];
  }
}

// ---------------------------------------------------------------- decode
__global__ __launch_bounds__(256) void k_decode(
    const float* __restrict__ S, const float* __restrict__ smoothed,
    const float* __restrict__ Mc, const float* __restrict__ apcr,
    float* __restrict__ out) {
  const int b = blockIdx.y;
  const int base = blockIdx.x * 256;
  const int tid = threadIdx.x;
  __shared__ float AL[256 * PP];
  __shared__ float McL[CC * PP];
  for (int idx = tid; idx < CC * PP; idx += 256) McL[idx] = Mc[idx];
  const float a = fminf(fmaxf(*apcr, 0.f), 1.f);
  const int n = base + tid;
  const int seg = n % NSEG;
  for (int p = 0; p < PP; ++p) {
    const float s = S[((size_t)b * NN + n) * PP + p];
    const float sm = smoothed[((size_t)b * NSEG + seg) * PP + p];
    AL[tid * PP + p] = a * sm + (1.f - a) * s;
  }
  __syncthreads();
  for (int c = 0; c < CC; ++c) {
    float acc = 0.f;
#pragma unroll
    for (int p = 0; p < PP; ++p) acc += AL[tid * PP + p] * McL[c * PP + p];
    out[((size_t)b * CC + c) * NN + base + tid] = acc;
  }
}

// ---------------------------------------------------------------- launch
extern "C" void kernel_launch(void* const* d_in, const int* in_sizes, int n_in,
                              void* d_out, int out_size, void* d_ws, size_t ws_size,
                              hipStream_t stream) {
  (void)in_sizes; (void)n_in; (void)out_size; (void)ws_size;
  const float* x   = (const float*)d_in[0];
  const float* c1w = (const float*)d_in[1];
  const float* c2w = (const float*)d_in[3];
  const float* c3w = (const float*)d_in[5];
  const float* c4w = (const float*)d_in[7];
  const float* c4b = (const float*)d_in[8];
  const float* g1W = (const float*)d_in[9];
  const float* g1a = (const float*)d_in[10];
  const float* g2W = (const float*)d_in[11];
  const float* g2a = (const float*)d_in[12];
  const float* fgw = (const float*)d_in[13];
  const float* fgb = (const float*)d_in[14];
  const float* flw = (const float*)d_in[15];
  const float* flb = (const float*)d_in[16];
  const float* alpha = (const float*)d_in[17];
  const float* beta  = (const float*)d_in[18];
  const float* apcr  = (const float*)d_in[19];
  const float* Mptr  = (const float*)d_in[20];
  float* out = (float*)d_out;

  char* ws = (char*)d_ws;
  size_t off = 0;
  auto alloc = [&](size_t bytes) -> void* {
    off = (off + 255) & ~(size_t)255;
    void* p = ws + off;
    off += bytes;
    return p;
  };
  __bf16* xT      = (__bf16*)alloc((size_t)BB * NN * CP * 2);
  float*  xnorm   = (float*)alloc((size_t)BB * NN * 4);
  __bf16* Q       = (__bf16*)alloc((size_t)BB * NN * KP * 2);
  __bf16* cbf     = (__bf16*)alloc((size_t)BB * KP * CP * 2);
  float*  cf32    = (float*)alloc((size_t)BB * KP * CP * 4);
  float*  cnorm   = (float*)alloc((size_t)BB * KP * 4);
  float*  csp     = (float*)alloc((size_t)BB * KP * 2 * 4);
  float*  mass    = (float*)alloc((size_t)BB * KP * 4);
  float*  cs_acc  = (float*)alloc((size_t)BB * KP * CP * 4);
  float*  csp_acc = (float*)alloc((size_t)BB * KP * 2 * 4);
  float*  gout    = (float*)alloc((size_t)BB * KP * PP * 4);
  float*  SG      = (float*)alloc((size_t)BB * NN * PP * 4);
  __bf16* wp1     = (__bf16*)alloc((size_t)64 * 9 * CP * 2);
  __bf16* wp2     = (__bf16*)alloc((size_t)128 * 9 * 64 * 2);
  __bf16* wp3     = (__bf16*)alloc((size_t)128 * 9 * 128 * 2);
  float*  conv    = (float*)alloc((size_t)BB * NN * 128 * 4);
  __bf16* t1      = (__bf16*)alloc((size_t)BB * NN * 64 * 2);
  __bf16* t2      = (__bf16*)alloc((size_t)BB * NN * 128 * 2);
  __bf16* t3      = (__bf16*)alloc((size_t)BB * NN * 128 * 2);
  float*  SC      = (float*)alloc((size_t)BB * NN * PP * 4);
  float*  Sbuf    = (float*)alloc((size_t)BB * NN * PP * 4);
  float*  spsum   = (float*)alloc((size_t)BB * NSEG * PP * 4);
  float*  smth    = (float*)alloc((size_t)BB * NSEG * PP * 4);
  float*  Mc      = (float*)alloc((size_t)CC * PP * 4);
  float*  ms      = (float*)alloc((size_t)BB * 128 * 2 * 4);

  // ---- global branch: DiffSLIC + GAT
  k_pack_x<<<dim3(NN / 256, BB), 256, 0, stream>>>(x, xT, xnorm);
  k_init_centers<<<1, 256, 0, stream>>>(x, cbf, cf32, csp);
  k_cnorm<<<1, 256, 0, stream>>>(cbf, cnorm);
  for (int it = 0; it < 5; ++it) {
    hipMemsetAsync(mass, 0, (size_t)BB * KP * 4, stream);
    hipMemsetAsync(csp_acc, 0, (size_t)BB * KP * 2 * 4, stream);
    hipMemsetAsync(cs_acc, 0, (size_t)BB * KP * CP * 4, stream);
    k_assign<<<dim3(NN / 128, BB), 256, ASSIGN_LDS, stream>>>(
        xT, xnorm, cbf, cnorm, csp, Q, mass, csp_acc);
    k_update<<<dim3(NN / UPD_PIX, BB), 256, UPD_LDS, stream>>>(Q, xT, cs_acc);
    k_finalize<<<BB * KP, 256, 0, stream>>>(cs_acc, csp_acc, mass, cbf, cf32, cnorm,
                                            csp);
  }
  k_gat<<<BB, 256, 0, stream>>>(cf32, g1W, g1a, g2W, g2a, gout);
  k_sg<<<dim3(NN / 256, BB), 256, 0, stream>>>(Q, gout, SG);

  // ---- local branch: CNN encoder (biases are zero; inst-norm cancels them)
  k_wpack<<<(64 * 9 * CP + 255) / 256, 256, 0, stream>>>(c1w, wp1, 64, CC, CP);
  k_conv<<<dim3(NN / 128, BB), 256, 0, stream>>>(xT, wp1, conv, CP, 64, 1);
  k_in_stats<<<dim3(64, BB), 256, 0, stream>>>(conv, ms, 64);
  k_in_apply<<<(int)(((size_t)BB * NN * 64) / 256), 256, 0, stream>>>(conv, ms, t1, 64);
  k_wpack<<<(128 * 9 * 64 + 255) / 256, 256, 0, stream>>>(c2w, wp2, 128, 64, 64);
  k_conv<<<dim3(NN / 128, BB), 256, 0, stream>>>(t1, wp2, conv, 64, 128, 2);
  k_in_stats<<<dim3(128, BB), 256, 0, stream>>>(conv, ms, 128);
  k_in_apply<<<(int)(((size_t)BB * NN * 128) / 256), 256, 0, stream>>>(conv, ms, t2, 128);
  k_wpack<<<(128 * 9 * 128 + 255) / 256, 256, 0, stream>>>(c3w, wp3, 128, 128, 128);
  k_conv<<<dim3(NN / 128, BB), 256, 0, stream>>>(t2, wp3, conv, 128, 128, 1);
  k_in_stats<<<dim3(128, BB), 256, 0, stream>>>(conv, ms, 128);
  k_in_apply<<<(int)(((size_t)BB * NN * 128) / 256), 256, 0, stream>>>(conv, ms, t3, 128);
  k_conv4<<<dim3(NN / 256, BB), 256, 0, stream>>>(t3, c4w, c4b, SC);

  // ---- fusion + PCR + decode
  hipMemsetAsync(spsum, 0, (size_t)BB * NSEG * PP * 4, stream);
  k_fusion<<<dim3(NN / 256, BB), 256, 0, stream>>>(SG, SC, fgw, fgb, flw, flb, alpha,
                                                   beta, Sbuf, spsum);
  k_smooth<<<BB, 256, 0, stream>>>(spsum, smth);
  k_mc<<<1, 256, 0, stream>>>(Mptr, Mc);
  k_decode<<<dim3(NN / 256, BB), 256, 0, stream>>>(Sbuf, smth, Mc, apcr, out);
}
